// QuantLinear_23424751632708
// MI455X (gfx1250) — matrix operand order repaired and checked by reading, not compiled, not run on hardware
//
#include <hip/hip_runtime.h>
#include <hip/hip_bf16.h>

// ---- vector types ----
typedef __attribute__((ext_vector_type(16))) __bf16 v16bf;
typedef __attribute__((ext_vector_type(8)))  __bf16 v8bf;
typedef __attribute__((ext_vector_type(8)))  float  v8f;
typedef __attribute__((ext_vector_type(4)))  float  v4f;
typedef __attribute__((ext_vector_type(4)))  int    v4i;

union FragBF { v16bf v; v8bf h[2]; };

// ---- problem size (compile-time) ----
constexpr int Mdim = 2 * 4096;   // B*S
constexpr int Ndim = 16384;      // OUT
constexpr int Kdim = 4096;       // IN

// ---- tiling ----
constexpr int BM = 128;
constexpr int BN = 256;
constexpr int BK = 64;
constexpr int LDA = BK + 8;      // padded LDS row stride (bf16 elems): 72 -> 36 banks, conflict-free b128
constexpr int LDB = BK + 8;
constexpr int THREADS = 256;     // 8 waves (wave32)

// =====================================================================
// Pre-conversion kernels (bandwidth-bound, run once per launch)
// =====================================================================
__global__ __launch_bounds__(256)
void cvt_x_kernel(const float* __restrict__ x, __bf16* __restrict__ o)
{
    const size_t i = ((size_t)blockIdx.x * 256 + threadIdx.x) * 8;
    v4f a = *(const v4f*)(x + i);
    v4f b = *(const v4f*)(x + i + 4);
    v8bf t;
    #pragma unroll
    for (int e = 0; e < 4; ++e) { t[e] = (__bf16)a[e]; t[4 + e] = (__bf16)b[e]; }
    *(v8bf*)(o + i) = t;
}

__global__ __launch_bounds__(256)
void cvt_w_kernel(const signed char* __restrict__ q, __bf16* __restrict__ o)
{
    const size_t i = ((size_t)blockIdx.x * 256 + threadIdx.x) * 16;
    v4i v = *(const v4i*)(q + i);
    const signed char* sb = (const signed char*)&v;
    v8bf t0, t1;
    #pragma unroll
    for (int e = 0; e < 8; ++e) {
        t0[e] = (__bf16)(float)sb[e];        // int8 exact in bf16
        t1[e] = (__bf16)(float)sb[8 + e];
    }
    *(v8bf*)(o + i)     = t0;
    *(v8bf*)(o + i + 8) = t1;
}

// =====================================================================
// Main GEMM: both operands bf16 in global; async DMA global->LDS,
// triple-buffered, one barrier per K-stage.
// =====================================================================
__global__ __launch_bounds__(THREADS, 1)
void qlinear_wmma_async(const __bf16* __restrict__ A,   // [Mdim,Kdim] bf16
                        const __bf16* __restrict__ Bw,  // [Ndim,Kdim] bf16
                        const float*  __restrict__ scale,
                        __bf16* __restrict__ out)
{
    // triple-buffered LDS: 3*(18KB + 36KB) = 166KB of the 320KB WGP budget
    __shared__ __align__(16) __bf16 ldsA[3][BM * LDA];
    __shared__ __align__(16) __bf16 ldsB[3][BN * LDB];

    const int tid   = threadIdx.x;
    const int lane  = tid & 31;
    const int wid   = tid >> 5;
    const int wm    = (wid >> 2) * 64;   // wave M offset: 0 / 64
    const int wn    = (wid & 3)  * 64;   // wave N offset: 0..192
    const int lmod  = lane & 15;
    const int lhalf = lane >> 4;

    const int m0 = blockIdx.y * BM;
    const int n0 = blockIdx.x * BN;

    // async-copy assignment (per K-stage slab): 16B granules per lane
    const int arow = tid >> 1;           // 0..127
    const int acol = (tid & 1) * 32;     // 0 / 32   (A: 4 x 16B per thread)
    const int brow = tid >> 2;           // 0..63    (B: 4 row-groups x 2 x 16B per thread)
    const int bcol = (tid & 3) * 16;

    // issue 12 async b128 copies: global (bf16) -> LDS buffer `buf`, K-slab s
    auto issue_async = [&](int buf, int s) {
        const int k0 = s * BK;
        {
            const __bf16* g = A + (size_t)(m0 + arow) * Kdim + k0 + acol;
            unsigned la = (unsigned)(uintptr_t)&ldsA[buf][arow * LDA + acol];
            unsigned long long ga = (unsigned long long)(uintptr_t)g;
            // same INST_OFFSET applies to both global and LDS sides; within a
            // row both advance 16B together, so immediate offsets are legal.
            asm volatile(
                "global_load_async_to_lds_b128 %0, %1, off\n\t"
                "global_load_async_to_lds_b128 %0, %1, off offset:16\n\t"
                "global_load_async_to_lds_b128 %0, %1, off offset:32\n\t"
                "global_load_async_to_lds_b128 %0, %1, off offset:48"
                :: "v"(la), "v"(ga) : "memory");
        }
        #pragma unroll
        for (int j = 0; j < 4; ++j) {
            const __bf16* g = Bw + (size_t)(n0 + j * 64 + brow) * Kdim + k0 + bcol;
            unsigned la = (unsigned)(uintptr_t)&ldsB[buf][(j * 64 + brow) * LDB + bcol];
            unsigned long long ga = (unsigned long long)(uintptr_t)g;
            asm volatile(
                "global_load_async_to_lds_b128 %0, %1, off\n\t"
                "global_load_async_to_lds_b128 %0, %1, off offset:16"
                :: "v"(la), "v"(ga) : "memory");
        }
    };

    v8f acc[4][4];
    #pragma unroll
    for (int i = 0; i < 4; ++i)
        #pragma unroll
        for (int j = 0; j < 4; ++j)
            acc[i][j] = v8f{};

    constexpr int nstage = Kdim / BK;    // 64

    // pipeline prologue: two slabs in flight
    issue_async(0, 0);
    issue_async(1, 1);

    int bc = 0, bn_ = 1, bi = 2;         // compute buf, next buf, issue target
    for (int s = 0; s < nstage; ++s) {
        // wait for this stage's 12 copies (async ops complete in order),
        // leaving the next stage's 12 still in flight.
        if (s + 1 < nstage) asm volatile("s_wait_asynccnt 0xc" ::: "memory");
        else                asm volatile("s_wait_asynccnt 0x0" ::: "memory");
        __syncthreads();    // single barrier/stage (triple buffering makes this safe)

        // ---- 2 x (4x4) WMMA k-steps from LDS[bc] ----
        #pragma unroll
        for (int kk = 0; kk < BK; kk += 32) {
            FragBF bfr[4];
            #pragma unroll
            for (int jn = 0; jn < 4; ++jn) {
                // B lane layout: lane = column n (mod 16); 16 consecutive K at kk + 16*lhalf
                const __bf16* p = &ldsB[bc][(wn + jn * 16 + lmod) * LDB + kk + lhalf * 16];
                bfr[jn].h[0] = *(const v8bf*)p;
                bfr[jn].h[1] = *(const v8bf*)(p + 8);
            }
            #pragma unroll
            for (int im = 0; im < 4; ++im) {
                // A lane layout: lane = row m (mod 16); 16 consecutive K at kk + 16*lhalf
                FragBF afr;
                const __bf16* p = &ldsA[bc][(wm + im * 16 + lmod) * LDA + kk + lhalf * 16];
                afr.h[0] = *(const v8bf*)p;
                afr.h[1] = *(const v8bf*)(p + 8);
                #pragma unroll
                for (int jn = 0; jn < 4; ++jn) {
                    acc[im][jn] = __builtin_amdgcn_wmma_f32_16x16x32_bf16(
                        false, afr.v, false, bfr[jn].v,
                        (short)0, acc[im][jn], false, false);
                }
            }
        }

        // refill the buffer read in the previous stage (free after this barrier)... two stages ahead
        if (s + 2 < nstage) issue_async(bi, s + 2);
        const int t = bc; bc = bn_; bn_ = bi; bi = t;   // rotate buffers
    }

    // ---- epilogue: per-output-channel scale, convert to bf16, store ----
    #pragma unroll
    for (int jn = 0; jn < 4; ++jn) {
        const int ng = n0 + wn + jn * 16 + lmod;
        const float sc = scale[ng];
        #pragma unroll
        for (int im = 0; im < 4; ++im) {
            const int mbase = m0 + wm + im * 16 + lhalf * 8;   // C/D: VGPR r -> M = r + 8*lhalf
            __bf16* op = out + (size_t)mbase * Ndim + ng;
            #pragma unroll
            for (int r = 0; r < 8; ++r)
                op[(size_t)r * Ndim] = (__bf16)(acc[im][jn][r] * sc);
        }
    }
}

// =====================================================================
// Fallback GEMM (round-2 version): convert-in-loop, used if d_ws is small
// =====================================================================
__global__ __launch_bounds__(THREADS, 1)
void qlinear_wmma_fallback(const float* __restrict__ x,
                           const signed char* __restrict__ qw,
                           const float* __restrict__ scale,
                           __bf16* __restrict__ out)
{
    __shared__ __align__(16) __bf16 ldsA[2][BM * LDA];
    __shared__ __align__(16) __bf16 ldsB[2][BN * LDB];

    const int tid   = threadIdx.x;
    const int lane  = tid & 31;
    const int wid   = tid >> 5;
    const int wm    = (wid >> 2) * 64;
    const int wn    = (wid & 3)  * 64;
    const int lmod  = lane & 15;
    const int lhalf = lane >> 4;

    const int m0 = blockIdx.y * BM;
    const int n0 = blockIdx.x * BN;

    const int arow = tid >> 1;
    const int acol = (tid & 1) * 32;
    const int brow = tid >> 2;
    const int bcol = (tid & 3) * 16;

    const float*       aptr = x  + (size_t)(m0 + arow) * Kdim + acol;
    const signed char* bptr = qw + (size_t)(n0 + brow) * Kdim + bcol;

    v4f areg[8];
    v4i breg[4];

    auto prefetch = [&](int s) {
        const int k0 = s * BK;
        #pragma unroll
        for (int i = 0; i < 8; ++i) areg[i] = *(const v4f*)(aptr + k0 + i * 4);
        #pragma unroll
        for (int j = 0; j < 4; ++j)
            breg[j] = *(const v4i*)(bptr + k0 + (size_t)(j * 64) * Kdim);
    };
    auto convert_store = [&](int buf) {
        __bf16* da = &ldsA[buf][arow * LDA + acol];
        #pragma unroll
        for (int c = 0; c < 4; ++c) {
            v8bf t;
            #pragma unroll
            for (int j2 = 0; j2 < 4; ++j2) {
                t[j2]     = (__bf16)areg[2 * c][j2];
                t[4 + j2] = (__bf16)areg[2 * c + 1][j2];
            }
            *(v8bf*)(da + c * 8) = t;
        }
        #pragma unroll
        for (int j = 0; j < 4; ++j) {
            __bf16* db = &ldsB[buf][(j * 64 + brow) * LDB + bcol];
            const signed char* sb = (const signed char*)&breg[j];
            #pragma unroll
            for (int c = 0; c < 2; ++c) {
                v8bf t;
                #pragma unroll
                for (int e = 0; e < 8; ++e)
                    t[e] = (__bf16)(float)sb[c * 8 + e];
                *(v8bf*)(db + c * 8) = t;
            }
        }
    };

    v8f acc[4][4];
    #pragma unroll
    for (int i = 0; i < 4; ++i)
        #pragma unroll
        for (int j = 0; j < 4; ++j)
            acc[i][j] = v8f{};

    constexpr int nstage = Kdim / BK;

    prefetch(0);
    convert_store(0);
    prefetch(1);

    for (int s = 0; s < nstage; ++s) {
        const int cur = s & 1;
        __syncthreads();
        #pragma unroll
        for (int kk = 0; kk < BK; kk += 32) {
            FragBF bfr[4];
            #pragma unroll
            for (int jn = 0; jn < 4; ++jn) {
                const __bf16* p = &ldsB[cur][(wn + jn * 16 + lmod) * LDB + kk + lhalf * 16];
                bfr[jn].h[0] = *(const v8bf*)p;
                bfr[jn].h[1] = *(const v8bf*)(p + 8);
            }
            #pragma unroll
            for (int im = 0; im < 4; ++im) {
                FragBF afr;
                const __bf16* p = &ldsA[cur][(wm + im * 16 + lmod) * LDA + kk + lhalf * 16];
                afr.h[0] = *(const v8bf*)p;
                afr.h[1] = *(const v8bf*)(p + 8);
                #pragma unroll
                for (int jn = 0; jn < 4; ++jn) {
                    acc[im][jn] = __builtin_amdgcn_wmma_f32_16x16x32_bf16(
                        false, afr.v, false, bfr[jn].v,
                        (short)0, acc[im][jn], false, false);
                }
            }
        }
        if (s + 1 < nstage) {
            convert_store(cur ^ 1);
            if (s + 2 < nstage) prefetch(s + 2);
        }
    }

    #pragma unroll
    for (int jn = 0; jn < 4; ++jn) {
        const int ng = n0 + wn + jn * 16 + lmod;
        const float sc = scale[ng];
        #pragma unroll
        for (int im = 0; im < 4; ++im) {
            const int mbase = m0 + wm + im * 16 + lhalf * 8;
            __bf16* op = out + (size_t)mbase * Ndim + ng;
            #pragma unroll
            for (int r = 0; r < 8; ++r)
                op[(size_t)r * Ndim] = (__bf16)(acc[im][jn][r] * sc);
        }
    }
}

// =====================================================================
extern "C" void kernel_launch(void* const* d_in, const int* in_sizes, int n_in,
                              void* d_out, int out_size, void* d_ws, size_t ws_size,
                              hipStream_t stream) {
    const float*       x     = (const float*)d_in[0];
    const signed char* qw    = (const signed char*)d_in[1];
    const float*       scale = (const float*)d_in[2];
    __bf16*            out   = (__bf16*)d_out;

    const size_t bytesA = (size_t)Mdim * Kdim * sizeof(__bf16);   //  67 MB
    const size_t bytesB = (size_t)Ndim * Kdim * sizeof(__bf16);   // 134 MB

    dim3 grid(Ndim / BN, Mdim / BM);   // 64 x 64 workgroups

    if (ws_size >= bytesA + bytesB) {
        __bf16* wsA = (__bf16*)d_ws;
        __bf16* wsB = (__bf16*)((char*)d_ws + bytesA);
        cvt_x_kernel<<<(int)((size_t)Mdim * Kdim / (256 * 8)),  256, 0, stream>>>(x, wsA);
        cvt_w_kernel<<<(int)((size_t)Ndim * Kdim / (256 * 16)), 256, 0, stream>>>(qw, wsB);
        qlinear_wmma_async<<<grid, THREADS, 0, stream>>>(wsA, wsB, scale, out);
    } else {
        qlinear_wmma_fallback<<<grid, THREADS, 0, stream>>>(x, qw, scale, out);
    }

    (void)in_sizes; (void)n_in; (void)out_size;
}
